// OptimalMatchingLayer_19713899889133
// MI455X (gfx1250) — matrix operand order repaired and checked by reading, not compiled, not run on hardware
//
#include <hip/hip_runtime.h>
#include <hip/hip_bf16.h>
#include <math.h>

typedef __attribute__((ext_vector_type(16))) _Float16 v16h;
typedef __attribute__((ext_vector_type(8)))  _Float16 v8h;
typedef __attribute__((ext_vector_type(8)))  float    v8f;

#define TAU_F 0.5f
#define ROI_R2 (0.1f * 0.1f)
#define NBINS 4096
#define MAX_SINK_ITERS 100

// ---------------------------------------------------------------- init
__global__ void init_kernel(unsigned int* __restrict__ hist, float* __restrict__ u,
                            float* __restrict__ v, int* __restrict__ ctrl,
                            float* __restrict__ errAcc, const int* __restrict__ iters,
                            int M, int N) {
  int i = blockIdx.x * blockDim.x + threadIdx.x;
  if (i < NBINS) hist[i] = 0u;
  if (i <= M) u[i] = 0.0f;
  if (i <= N) v[i] = 0.0f;
  if (i == 0) {
    ctrl[0] = (iters[0] <= 0) ? 1 : 0;  // done flag
    ctrl[1] = 0;                        // iteration counter
    *errAcc = 0.0f;
  }
}

// ------------------------------------------------- normalize rows -> f16
__global__ void normalize_f16_kernel(const float* __restrict__ e,
                                     _Float16* __restrict__ out, int n, int d) {
  int row = blockIdx.x;
  int t = threadIdx.x;
  __shared__ float red[256];
  const float* src = e + (size_t)row * d;
  float ss = 0.0f;
  for (int j = t; j < d; j += blockDim.x) { float x = src[j]; ss += x * x; }
  red[t] = ss; __syncthreads();
  for (int s = 128; s > 0; s >>= 1) { if (t < s) red[t] += red[t + s]; __syncthreads(); }
  float inv = 1.0f / fmaxf(sqrtf(red[0]), 1e-12f);
  _Float16* dst = out + (size_t)row * d;
  for (int j = t; j < d; j += blockDim.x) dst[j] = (_Float16)(src[j] * inv);
}

// -------------------------------------- WMMA GEMM + ROI mask -> couplings
// Block = 256 threads = 8 waves; each wave computes one 16x16 C-tile.
// Block tile = 32 rows x 64 cols. K in steps of 32 (v_wmma_f32_16x16x32_f16).
__global__ void wmma_scores_kernel(const _Float16* __restrict__ A,  // n1 x d row-major, normalized
                                   const _Float16* __restrict__ B,  // n2 x d row-major, normalized
                                   const float* __restrict__ loc1,
                                   const float* __restrict__ loc2,
                                   float* __restrict__ C, int n1, int n2, int d, int ldc) {
  const int wave = threadIdx.x >> 5;
  const int lane = threadIdx.x & 31;
  const int row0 = (blockIdx.y * 2 + (wave >> 2)) * 16;
  const int col0 = (blockIdx.x * 4 + (wave & 3)) * 16;
  const int g  = lane >> 4;   // half-wave group
  const int mr = lane & 15;

  const _Float16* arow = A + (size_t)(row0 + mr) * d;  // A fragment: row = M
  const _Float16* brow = B + (size_t)(col0 + mr) * d;  // B fragment: row = N (K contiguous)

  v8f acc = {};
  for (int k0 = 0; k0 < d; k0 += 32) {
    if (k0 + 32 < d) {  // global_prefetch_b8 for next K-slice
      __builtin_prefetch(arow + k0 + 32, 0, 1);
      __builtin_prefetch(brow + k0 + 32, 0, 1);
    }
    // 16-bit A 16x32 layout: lanes 0-15 hold K=0..7 / 16..23, lanes 16-31 K=8..15 / 24..31
    v8h a_lo = *(const v8h*)(arow + k0 + g * 8);
    v8h a_hi = *(const v8h*)(arow + k0 + 16 + g * 8);
    // B 32x16 layout: lanes 0-15 hold K=0..7 / 16..23, lanes 16-31 K=8..15 / 24..31
    v8h b_lo = *(const v8h*)(brow + k0 + g * 8);
    v8h b_hi = *(const v8h*)(brow + k0 + 16 + g * 8);
    v16h a, b;
#pragma unroll
    for (int h = 0; h < 8; ++h) {
      a[h] = a_lo[h]; a[8 + h] = a_hi[h];
      b[h] = b_lo[h]; b[8 + h] = b_hi[h];
    }
    acc = __builtin_amdgcn_wmma_f32_16x16x32_f16(false, a, false, b, (short)0, acc,
                                                 false, false);
  }

  // Epilogue: C/D layout — lanes 0-15: N=lane, VGPR j -> M=j; lanes 16-31: M=j+8
  const int n_local = lane & 15;
  const int cc = col0 + n_local;
  const float l2x = loc2[cc * 2 + 0], l2y = loc2[cc * 2 + 1];
  const float inv_denom = 1.0f / (2.0f * TAU_F + 1e-6f);
#pragma unroll
  for (int j = 0; j < 8; ++j) {
    int r = row0 + j + g * 8;
    float dx = loc1[r * 2 + 0] - l2x;
    float dy = loc1[r * 2 + 1] - l2y;
    float val = 0.0f;
    if (dx * dx + dy * dy < ROI_R2) val = (acc[j] + 1.0f) * inv_denom;
    C[(size_t)r * ldc + cc] = val;
  }
}

// ---------------------------------------------- histogram of nonzero scores
__global__ void hist_kernel(const float* __restrict__ C, int n1, int n2, int ldc,
                            unsigned int* __restrict__ hist) {
  __shared__ unsigned int h[NBINS];
  for (int i = threadIdx.x; i < NBINS; i += blockDim.x) h[i] = 0u;
  __syncthreads();
  size_t total = (size_t)n1 * n2;
  size_t stride = (size_t)gridDim.x * blockDim.x;
  for (size_t idx = (size_t)blockIdx.x * blockDim.x + threadIdx.x; idx < total; idx += stride) {
    size_t r = idx / n2, c = idx % n2;
    float v = C[r * ldc + c];
    if (v != 0.0f) {
      int b = (int)(v * (NBINS * 0.5f));  // scores lie in [0, 2)
      b = min(max(b, 0), NBINS - 1);
      atomicAdd(&h[b], 1u);
    }
  }
  __syncthreads();
  for (int i = threadIdx.x; i < NBINS; i += blockDim.x)
    if (h[i]) atomicAdd(&hist[i], h[i]);
}

__global__ void median_kernel(const unsigned int* __restrict__ hist,
                              float* __restrict__ binScore) {
  if (threadIdx.x == 0) {
    unsigned long long nnz = 0;
    for (int i = 0; i < NBINS; ++i) nnz += hist[i];
    unsigned long long target = (nnz == 0) ? 0ull : ((nnz - 1ull) >> 1);
    unsigned long long cum = 0; int bin = 0;
    for (int i = 0; i < NBINS; ++i) { cum += hist[i]; if (cum > target) { bin = i; break; } }
    *binScore = (bin + 0.5f) * (2.0f / NBINS);
  }
}

// ----------------------------------- pad couplings (last row / last col / corner)
__global__ void pad_kernel(float* __restrict__ C, int n1, int n2, int ldc,
                           const float* __restrict__ binScore) {
  float b = *binScore;
  int i = blockIdx.x * blockDim.x + threadIdx.x;
  if (i <= n2) C[(size_t)n1 * ldc + i] = b;  // bottom row incl. corner
  if (i <  n1) C[(size_t)i * ldc + n2] = b;  // right column
}

// --------------------------------------------------- Sinkhorn: u update (rows)
__global__ void sink_row_kernel(const float* __restrict__ C, const float* __restrict__ v,
                                float* __restrict__ u, float* __restrict__ errAcc,
                                const int* __restrict__ ctrl, int M, int N, int ldc,
                                float lmu_main, float lmu_last, float invEps) {
  if (ctrl[0]) return;
  int i = blockIdx.x;          // 0..M inclusive
  int t = threadIdx.x;
  __shared__ float sm[256], ss[256];
  const float* row = C + (size_t)i * ldc;
  float mx = -INFINITY, sum = 0.0f;
  for (int j = t; j <= N; j += blockDim.x) {
    float x = row[j] * invEps + v[j];
    if (x > mx) { sum = sum * __expf(mx - x) + 1.0f; mx = x; }
    else        { sum += __expf(x - mx); }
  }
  sm[t] = mx; ss[t] = sum; __syncthreads();
  for (int s = 128; s > 0; s >>= 1) {
    if (t < s) {
      float m2 = sm[t + s], s2 = ss[t + s];
      if (m2 > sm[t]) { ss[t] = ss[t] * __expf(sm[t] - m2) + s2; sm[t] = m2; }
      else if (m2 != -INFINITY) { ss[t] += s2 * __expf(m2 - sm[t]); }
    }
    __syncthreads();
  }
  if (t == 0) {
    float lse = sm[0] + __logf(ss[0]);
    float un = ((i < M) ? lmu_main : lmu_last) - lse;
    float old = u[i];
    u[i] = un;
    atomicAdd(errAcc, fabsf(un - old));
  }
}

// --------------------------------------------------- Sinkhorn: v update (cols)
// 256 threads = 32 cols x 8 row-strips; consecutive lanes read consecutive cols.
__global__ void sink_col_kernel(const float* __restrict__ C, const float* __restrict__ u,
                                float* __restrict__ v, const int* __restrict__ ctrl,
                                int M, int N, int ldc,
                                float lnu_main, float lnu_last, float invEps) {
  if (ctrl[0]) return;
  int tx = threadIdx.x & 31, ty = threadIdx.x >> 5;
  int col = blockIdx.x * 32 + tx;
  __shared__ float sm[8][33], ss[8][33];
  float mx = -INFINITY, sum = 0.0f;
  if (col <= N) {
    for (int i = ty; i <= M; i += 8) {
      float x = C[(size_t)i * ldc + col] * invEps + u[i];
      if (x > mx) { sum = sum * __expf(mx - x) + 1.0f; mx = x; }
      else        { sum += __expf(x - mx); }
    }
  }
  sm[ty][tx] = mx; ss[ty][tx] = sum; __syncthreads();
  if (ty == 0 && col <= N) {
    float M0 = sm[0][tx], S0 = ss[0][tx];
    for (int r = 1; r < 8; ++r) {
      float m2 = sm[r][tx], s2 = ss[r][tx];
      if (m2 > M0) { S0 = S0 * __expf(M0 - m2) + s2; M0 = m2; }
      else if (m2 != -INFINITY) { S0 += s2 * __expf(m2 - M0); }
    }
    v[col] = ((col < N) ? lnu_main : lnu_last) - (M0 + __logf(S0));
  }
}

__global__ void sink_check_kernel(int* __restrict__ ctrl, float* __restrict__ errAcc,
                                  const int* __restrict__ iters) {
  if (threadIdx.x == 0 && !ctrl[0]) {
    int it = ++ctrl[1];
    float err = *errAcc;
    if (it >= iters[0] || err < 1e-4f) ctrl[0] = 1;
    *errAcc = 0.0f;
  }
}

// ------------------------------------------- final soft = C + u + v - norm
__global__ void soft_kernel(float* __restrict__ C, const float* __restrict__ u,
                            const float* __restrict__ v, int M, int N, int ldc,
                            float norm) {
  int j = blockIdx.x * blockDim.x + threadIdx.x;
  int i = blockIdx.y;
  if (j <= N) C[(size_t)i * ldc + j] += u[i] + v[j] - norm;
}

// --------------------------------------------------------------- top-2
struct Top2 { float v0, v1; int i0, i1; };

__device__ inline void t2_push(Top2& t, float val, int idx) {
  if (val > t.v0 || (val == t.v0 && idx < t.i0)) {
    t.v1 = t.v0; t.i1 = t.i0; t.v0 = val; t.i0 = idx;
  } else if (val > t.v1 || (val == t.v1 && idx < t.i1)) {
    t.v1 = val; t.i1 = idx;
  }
}

__global__ void top2_row_kernel(const float* __restrict__ S, int M, int N, int ldc,
                                float* __restrict__ vals0, int* __restrict__ idx0) {
  int i = blockIdx.x, t = threadIdx.x;
  __shared__ Top2 red[256];
  Top2 loc = { -INFINITY, -INFINITY, 0, 0 };
  const float* row = S + (size_t)i * ldc;
  for (int j = t; j < N; j += blockDim.x) t2_push(loc, row[j], j);
  red[t] = loc; __syncthreads();
  for (int s = 128; s > 0; s >>= 1) {
    if (t < s) { Top2 o = red[t + s]; t2_push(red[t], o.v0, o.i0); t2_push(red[t], o.v1, o.i1); }
    __syncthreads();
  }
  if (t == 0) {
    vals0[0 * M + i] = red[0].v0; vals0[1 * M + i] = red[0].v1;
    idx0 [0 * M + i] = red[0].i0; idx0 [1 * M + i] = red[0].i1;
  }
}

__global__ void top2_col_kernel(const float* __restrict__ S, int M, int N, int ldc,
                                float* __restrict__ vals1, int* __restrict__ idx1) {
  int tx = threadIdx.x & 31, ty = threadIdx.x >> 5;
  int col = blockIdx.x * 32 + tx;
  __shared__ Top2 red[8][33];
  Top2 loc = { -INFINITY, -INFINITY, 0, 0 };
  if (col < N) for (int i = ty; i < M; i += 8) t2_push(loc, S[(size_t)i * ldc + col], i);
  red[ty][tx] = loc; __syncthreads();
  if (ty == 0 && col < N) {
    Top2 a = red[0][tx];
    for (int r = 1; r < 8; ++r) { Top2 o = red[r][tx]; t2_push(a, o.v0, o.i0); t2_push(a, o.v1, o.i1); }
    vals1[0 * N + col] = a.v0; vals1[1 * N + col] = a.v1;
    idx1 [0 * N + col] = a.i0; idx1 [1 * N + col] = a.i1;
  }
}

// ------------------------------------------------------- mutual matching
__global__ void match0_kernel(const float* __restrict__ vals0, const int* __restrict__ idx0,
                              const int* __restrict__ idx1, int M, int N,
                              float* __restrict__ mscore0, float* __restrict__ ind0_out,
                              int* __restrict__ valid0) {
  int m = blockIdx.x * blockDim.x + threadIdx.x;
  if (m >= M) return;
  float acc = 0.0f;
#pragma unroll
  for (int k = 0; k < 2; ++k) {
    int i0 = idx0[k * M + m];                 // chosen column
    bool mutual = (idx1[k * N + i0] == m);
    float ms = mutual ? __expf(vals0[k * M + m]) : 0.0f;
    acc += ms;
    bool valid = mutual && (ms > 0.0f);
    valid0[k * M + m] = valid ? 1 : 0;
    ind0_out[k * M + m] = valid ? (float)i0 : -1.0f;
  }
  mscore0[m] = acc;
}

__global__ void match1_kernel(const float* __restrict__ vals1, const int* __restrict__ idx1,
                              const int* __restrict__ idx0, const int* __restrict__ valid0,
                              int M, int N, float* __restrict__ mscore1,
                              float* __restrict__ ind1_out) {
  int n = blockIdx.x * blockDim.x + threadIdx.x;
  if (n >= N) return;
  float acc = 0.0f;
#pragma unroll
  for (int k = 0; k < 2; ++k) {
    int i1 = idx1[k * N + n];                 // chosen row
    bool mutual = (idx0[k * M + i1] == n);
    float ms = mutual ? __expf(vals1[k * N + n]) : 0.0f;
    acc += ms;
    bool valid = mutual && (valid0[k * M + i1] != 0);
    ind1_out[k * N + n] = valid ? (float)i1 : -1.0f;
  }
  mscore1[n] = acc;
}

static inline int imax3(int a, int b, int c) { int m = a > b ? a : b; return m > c ? m : c; }

extern "C" void kernel_launch(void* const* d_in, const int* in_sizes, int n_in,
                              void* d_out, int out_size, void* d_ws, size_t ws_size,
                              hipStream_t stream) {
  const float* e1 = (const float*)d_in[0];
  const float* e2 = (const float*)d_in[1];
  const float* l1 = (const float*)d_in[2];
  const float* l2 = (const float*)d_in[3];
  const int* iters = (const int*)d_in[4];

  const int M = in_sizes[2] / 2;          // 4096
  const int N = in_sizes[3] / 2;          // 4096
  const int D = in_sizes[0] / M;          // 256
  const int ldc = N + 1;                  // 4097

  float* C = (float*)d_out;               // couplings live in-place in `soft`
  size_t softElems = (size_t)(M + 1) * (N + 1);
  float* mscore0 = C + softElems;
  float* mscore1 = mscore0 + M;
  float* ind0    = mscore1 + N;           // indices0 as float (2 x M)
  float* ind1    = ind0 + 2 * (size_t)M;  // indices1 as float (2 x N)

  // workspace carve-out (256B aligned blocks)
  char* ws = (char*)d_ws;
  size_t off = 0;
  auto carve = [&](size_t bytes) { size_t o = off; off = (off + bytes + 255) & ~(size_t)255; return o; };
  _Float16* A16   = (_Float16*)(ws + carve((size_t)M * D * sizeof(_Float16)));
  _Float16* B16   = (_Float16*)(ws + carve((size_t)N * D * sizeof(_Float16)));
  float* u        = (float*)(ws + carve((size_t)(M + 1) * 4));
  float* v        = (float*)(ws + carve((size_t)(N + 1) * 4));
  unsigned int* hist = (unsigned int*)(ws + carve((size_t)NBINS * 4));
  float* binScore = (float*)(ws + carve(4));
  int* ctrl       = (int*)(ws + carve(8));
  float* errAcc   = (float*)(ws + carve(4));
  float* vals0    = (float*)(ws + carve(2 * (size_t)M * 4));
  int*   idx0     = (int*)  (ws + carve(2 * (size_t)M * 4));
  float* vals1    = (float*)(ws + carve(2 * (size_t)N * 4));
  int*   idx1     = (int*)  (ws + carve(2 * (size_t)N * 4));
  int*   valid0   = (int*)  (ws + carve(2 * (size_t)M * 4));
  (void)ws_size; (void)n_in; (void)out_size;

  const float ms = (float)M, ns = (float)N;
  const float norm = -logf(ms + ns);
  const float lmu_main = norm, lmu_last = logf(ns) + norm;
  const float lnu_main = norm, lnu_last = logf(ms) + norm;
  const float invEps = 1.0f / (1.0f + 1e-6f);

  int initN = imax3(NBINS, M + 1, N + 1);
  init_kernel<<<(initN + 255) / 256, 256, 0, stream>>>(hist, u, v, ctrl, errAcc, iters, M, N);

  normalize_f16_kernel<<<M, 256, 0, stream>>>(e1, A16, M, D);
  normalize_f16_kernel<<<N, 256, 0, stream>>>(e2, B16, N, D);

  dim3 gemmGrid(N / 64, M / 32);
  wmma_scores_kernel<<<gemmGrid, 256, 0, stream>>>(A16, B16, l1, l2, C, M, N, D, ldc);

  hist_kernel<<<1024, 256, 0, stream>>>(C, M, N, ldc, hist);
  median_kernel<<<1, 64, 0, stream>>>(hist, binScore);

  int padN = (M > N + 1 ? M : N + 1);
  pad_kernel<<<(padN + 255) / 256, 256, 0, stream>>>(C, M, N, ldc, binScore);

  // Fixed launch count; device-side `ctrl` flag makes extra iterations no-ops
  // (emulates `while (i < iters && err >= 1e-4)` capture-safely).
  for (int it = 0; it < MAX_SINK_ITERS; ++it) {
    sink_row_kernel<<<M + 1, 256, 0, stream>>>(C, v, u, errAcc, ctrl, M, N, ldc,
                                               lmu_main, lmu_last, invEps);
    sink_col_kernel<<<(N + 1 + 31) / 32, 256, 0, stream>>>(C, u, v, ctrl, M, N, ldc,
                                                           lnu_main, lnu_last, invEps);
    sink_check_kernel<<<1, 32, 0, stream>>>(ctrl, errAcc, iters);
  }

  soft_kernel<<<dim3((N + 1 + 255) / 256, M + 1), 256, 0, stream>>>(C, u, v, M, N, ldc, norm);

  top2_row_kernel<<<M, 256, 0, stream>>>(C, M, N, ldc, vals0, idx0);
  top2_col_kernel<<<(N + 31) / 32, 256, 0, stream>>>(C, M, N, ldc, vals1, idx1);
  match0_kernel<<<(M + 255) / 256, 256, 0, stream>>>(vals0, idx0, idx1, M, N, mscore0, ind0, valid0);
  match1_kernel<<<(N + 255) / 256, 256, 0, stream>>>(vals1, idx1, idx0, valid0, M, N, mscore1, ind1);
}
